// multi_head_attention_2d_5669356832203
// MI455X (gfx1250) — matrix operand order repaired and checked by reading, not compiled, not run on hardware
//
#include <hip/hip_runtime.h>
#include <hip/hip_bf16.h>
#include <stdint.h>

// ---------------------------------------------------------------------------
// MI455X (gfx1250) multi-head global attention.
//  - f16 WMMA (v_wmma_f32_16x16x32_f16), f32 accumulate
//  - flash-attention (online softmax), 64-key LDS tiles
//  - double-buffered GLOBAL_LOAD_ASYNC_TO_LDS_B128 K/V staging (ASYNCcnt)
//  - wave32, 16x16 C/D tiles (8 VGPRs)
// ---------------------------------------------------------------------------

typedef _Float16 v16h __attribute__((ext_vector_type(16)));
typedef _Float16 v8h  __attribute__((ext_vector_type(8)));
typedef float    v8f  __attribute__((ext_vector_type(8)));
typedef int      v4i  __attribute__((ext_vector_type(4)));

typedef __attribute__((address_space(1))) v4i* gptr_v4i;   // global
typedef __attribute__((address_space(3))) v4i* lptr_v4i;   // LDS

__device__ __forceinline__ v8f wmma16x16x32(v16h a, v16h b, v8f c) {
    // (neg_a, A, neg_b, B, c_mod, C, reuse_a, reuse_b)
    return __builtin_amdgcn_wmma_f32_16x16x32_f16(false, a, false, b, (short)0, c,
                                                  false, false);
}

#if defined(__HIP_DEVICE_COMPILE__) && defined(__has_builtin)
#if __has_builtin(__builtin_amdgcn_global_load_async_to_lds_b128) && \
    __has_builtin(__builtin_amdgcn_s_wait_asynccnt)
#define HAVE_ASYNC_LDS 1
#endif
#endif
#ifndef HAVE_ASYNC_LDS
#define HAVE_ASYNC_LDS 0
#endif

#define N_SPATIAL 2048   // B*H*W = 2*32*32
#define C_DIM     256
#define HW        1024
#define NROWS     8192   // N_SPATIAL * NUM_HEADS
#define DHEAD     64
#define NKT       (NROWS / 64)   // 128 K/V tiles

// ---------------------------------------------------------------------------
// Kernel 1: fused QKV projection.
//   q[n,o] = sum_c x[b,c,s] * W[o,c] + bias[o]   (Q additionally * 1/8)
//   stored as f16 [2048][256]  ==  [8192][64] head-flattened.
// Grid: (32 row-tiles, 4 col-tiles, 3 matrices), block 256 (8 waves).
// ---------------------------------------------------------------------------
__global__ __launch_bounds__(256) void proj_qkv_kernel(
    const float* __restrict__ x,
    const float* __restrict__ Wq, const float* __restrict__ bq,
    const float* __restrict__ Wk, const float* __restrict__ bk,
    const float* __restrict__ Wv, const float* __restrict__ bv,
    _Float16* __restrict__ Qws, _Float16* __restrict__ Kws,
    _Float16* __restrict__ Vws)
{
    __shared__ _Float16 Xs[64][72];   // [row n][k]   (pad: 72 halves = 144B)
    __shared__ _Float16 Wsh[64][72];  // [col o][k]

    const int rt  = blockIdx.x;           // row tile (64 spatial positions)
    const int ct  = blockIdx.y;           // col tile (64 output channels)
    const int mat = blockIdx.z;           // 0=Q 1=K 2=V

    const float* W    = (mat == 0) ? Wq : (mat == 1) ? Wk : Wv;
    const float* bias = (mat == 0) ? bq : (mat == 1) ? bk : bv;
    _Float16*    outp = (mat == 0) ? Qws : (mat == 1) ? Kws : Vws;
    const float  sc   = (mat == 0) ? 0.125f : 1.0f;   // 1/sqrt(dk)

    const int n0 = rt * 64;
    const int o0 = ct * 64;
    const int b  = n0 >> 10;          // 64-row tiles never straddle a batch
    const int s0 = n0 & 1023;

    const int tid  = threadIdx.x;
    const int lane = tid & 31;
    const int wave = tid >> 5;        // 0..7
    const int hi   = lane >> 4;
    const int ln   = lane & 15;
    const int mrow = (wave & 3) * 16; // wave's 16-row stripe inside 64
    const int nt0  = (wave >> 2) * 2; // wave's first 16-col tile (0 or 2)

    v8f acc0 = {}; v8f acc1 = {};

    for (int kc = 0; kc < C_DIM; kc += 64) {
        __syncthreads();
        {   // X chunk: Xs[i][j] = x[b][kc+j][s0+i]
            // thread -> contiguous 16-half LDS run; global reads coalesced over i
            const int i  = tid & 63;
            const int jb = (tid >> 6) * 16;
            #pragma unroll
            for (int q = 0; q < 16; ++q)
                Xs[i][jb + q] =
                    (_Float16)x[((size_t)b * C_DIM + kc + jb + q) * HW + s0 + i];
            // W chunk: Wsh[i][j] = W[o0+i][kc+j]  (contiguous both sides)
            const int iw = tid >> 2;
            const int jw = (tid & 3) * 16;
            const float* wsrc = W + (size_t)(o0 + iw) * C_DIM + kc + jw;
            #pragma unroll
            for (int q = 0; q < 16; ++q) Wsh[iw][jw + q] = (_Float16)wsrc[q];
        }
        __syncthreads();
        #pragma unroll
        for (int ks = 0; ks < 2; ++ks) {
            // A fragment: 16x32 f16, lane = row, contiguous k = 16*hi + p
            v16h a;
            const int arow = mrow + ln;
            const int kb   = ks * 32 + 16 * hi;
            #pragma unroll
            for (int p = 0; p < 8; ++p) {
                a[p]     = Xs[arow][kb + p];
                a[8 + p] = Xs[arow][kb + 8 + p];
            }
            // B fragments: 32x16 f16 (B[k][n] = W[n][k]), lane = col
            v16h b0, b1;
            const int c0  = nt0 * 16 + ln;
            const int kb2 = ks * 32 + 16 * hi;
            #pragma unroll
            for (int p = 0; p < 16; ++p) {
                b0[p] = Wsh[c0][kb2 + p];
                b1[p] = Wsh[c0 + 16][kb2 + p];
            }
            acc0 = wmma16x16x32(a, b0, acc0);
            acc1 = wmma16x16x32(a, b1, acc1);
        }
    }
    // Epilogue: + bias, * scale, store f16.
    const int oc0 = o0 + nt0 * 16 + ln;
    const float bi0 = bias[oc0], bi1 = bias[oc0 + 16];
    #pragma unroll
    for (int r = 0; r < 8; ++r) {
        const int n = n0 + mrow + r + 8 * hi;      // C/D layout: M = r + 8*hi
        outp[(size_t)n * C_DIM + oc0]      = (_Float16)((acc0[r] + bi0) * sc);
        outp[(size_t)n * C_DIM + oc0 + 16] = (_Float16)((acc1[r] + bi1) * sc);
    }
}

// ---------------------------------------------------------------------------
// Kernel 2: flash attention over the globally flattened [8192, 64] Q/K/V.
// Block = 128 threads (4 waves), each wave owns 16 query rows -> 64/WG.
// K/V streamed in 64-key LDS tiles, double-buffered via async-to-LDS copies.
// ---------------------------------------------------------------------------
__device__ __forceinline__ void issue_kv_tile(
    const _Float16* __restrict__ K, const _Float16* __restrict__ V, int t,
    _Float16 (*Ksb)[72], _Float16 (*Vsb)[72], int tid)
{
#if HAVE_ASYNC_LDS
    // 64 rows x 128B per tile = 512 x b128 chunks; 128 threads -> 4 each.
    #pragma unroll
    for (int e = 0; e < 4; ++e) {
        const int idx = tid + e * 128;            // 0..511
        const int row = idx >> 3, c = (idx & 7) * 8;
        const _Float16* gk = K + ((size_t)t * 64 + row) * DHEAD + c;
        const _Float16* gv = V + ((size_t)t * 64 + row) * DHEAD + c;
        __builtin_amdgcn_global_load_async_to_lds_b128(
            (gptr_v4i)gk, (lptr_v4i)&Ksb[row][c], 0, 0);
        __builtin_amdgcn_global_load_async_to_lds_b128(
            (gptr_v4i)gv, (lptr_v4i)&Vsb[row][c], 0, 0);
    }
#else
    const uint32_t* ksrc = (const uint32_t*)(K + (size_t)t * 64 * DHEAD);
    const uint32_t* vsrc = (const uint32_t*)(V + (size_t)t * 64 * DHEAD);
    #pragma unroll
    for (int e = 0; e < 16; ++e) {
        const int idx = tid + e * 128;            // 0..2047 dwords
        const int row = idx >> 5, col = idx & 31;
        ((uint32_t*)&Ksb[row][0])[col] = ksrc[idx];
        ((uint32_t*)&Vsb[row][0])[col] = vsrc[idx];
    }
#endif
}

__global__ __launch_bounds__(128) void flash_attn_kernel(
    const _Float16* __restrict__ Q, const _Float16* __restrict__ K,
    const _Float16* __restrict__ V, _Float16* __restrict__ O)
{
    __shared__ _Float16 Ks[2][64][72];     // [buf][key][d]
    __shared__ _Float16 Vs[2][64][72];     // [buf][key][d]
    __shared__ _Float16 Ps[4][16][72];     // per-wave P re-stripe buffer

    const int tid  = threadIdx.x;
    const int lane = tid & 31;
    const int wave = tid >> 5;             // 0..3
    const int hi   = lane >> 4;
    const int ln   = lane & 15;
    const int qbase = blockIdx.x * 64 + wave * 16;

    // Resident Q A-fragments (16 rows x 64 deep = 2 x 16x32), b128 loads.
    v16h qa0, qa1;
    {
        const v8h* qv = (const v8h*)(Q + (size_t)(qbase + ln) * DHEAD);
        v8h q0 = qv[2 * hi], q1 = qv[2 * hi + 1], q2 = qv[4 + 2 * hi], q3 = qv[5 + 2 * hi];
        qa0 = __builtin_shufflevector(q0, q1, 0, 1, 2, 3, 4, 5, 6, 7,
                                      8, 9, 10, 11, 12, 13, 14, 15);
        qa1 = __builtin_shufflevector(q2, q3, 0, 1, 2, 3, 4, 5, 6, 7,
                                      8, 9, 10, 11, 12, 13, 14, 15);
    }

    v8f oacc[4] = {v8f{}, v8f{}, v8f{}, v8f{}};
    float m[8], lsum[8];
    #pragma unroll
    for (int r = 0; r < 8; ++r) { m[r] = -3.0e30f; lsum[r] = 0.0f; }

    issue_kv_tile(K, V, 0, Ks[0], Vs[0], tid);

    for (int t = 0; t < NKT; ++t) {
        const int cur = t & 1;
#if HAVE_ASYNC_LDS
        __builtin_amdgcn_s_wait_asynccnt(0);   // own tile-t copies landed
#endif
        __syncthreads();                        // whole tile visible; prev buf free
        if (t + 1 < NKT)                        // overlap t+1 copies with compute
            issue_kv_tile(K, V, t + 1, Ks[1 - cur], Vs[1 - cur], tid);

        // ---- S = Q @ K^T : 4 column tiles of 16 keys -----------------------
        v8f s[4];
        #pragma unroll
        for (int nt = 0; nt < 4; ++nt) {
            v16h b0, b1;
            const int krow = nt * 16 + ln;   // B lane = key column
            const int kb   = 16 * hi;
            #pragma unroll
            for (int p = 0; p < 16; ++p) {
                b0[p] = Ks[cur][krow][kb + p];        // K-dim 0..31
                b1[p] = Ks[cur][krow][32 + kb + p];   // K-dim 32..63
            }
            v8f c = {};
            c = wmma16x16x32(qa0, b0, c);
            c = wmma16x16x32(qa1, b1, c);
            s[nt] = c;
        }

        // ---- online softmax (rows 0-7 in lanes 0-15, rows 8-15 in 16-31) ---
        float mnew[8];
        #pragma unroll
        for (int r = 0; r < 8; ++r) {
            float mx = fmaxf(fmaxf(s[0][r], s[1][r]), fmaxf(s[2][r], s[3][r]));
            #pragma unroll
            for (int off = 1; off < 16; off <<= 1)
                mx = fmaxf(mx, __shfl_xor(mx, off, 32));
            mnew[r] = fmaxf(m[r], mx);
            const float corr = __expf(m[r] - mnew[r]);
            lsum[r] *= corr;
            oacc[0][r] *= corr; oacc[1][r] *= corr;
            oacc[2][r] *= corr; oacc[3][r] *= corr;
            m[r] = mnew[r];
        }
        #pragma unroll
        for (int nt = 0; nt < 4; ++nt) {
            #pragma unroll
            for (int r = 0; r < 8; ++r) {
                const float p = __expf(s[nt][r] - mnew[r]);
                s[nt][r] = p;
                Ps[wave][r + 8 * hi][nt * 16 + ln] = (_Float16)p;
            }
        }
        #pragma unroll
        for (int r = 0; r < 8; ++r) {
            float ssum = (s[0][r] + s[1][r]) + (s[2][r] + s[3][r]);
            #pragma unroll
            for (int off = 1; off < 16; off <<= 1)
                ssum += __shfl_xor(ssum, off, 32);
            lsum[r] += ssum;
        }

        // ---- O += P @ V ----------------------------------------------------
        v16h pa0, pa1;   // re-striped P as A fragments (LDS round trip)
        {
            const int kb = 16 * hi;
            #pragma unroll
            for (int p = 0; p < 8; ++p) {
                pa0[p]     = Ps[wave][ln][kb + p];
                pa0[8 + p] = Ps[wave][ln][kb + 8 + p];
                pa1[p]     = Ps[wave][ln][32 + kb + p];
                pa1[8 + p] = Ps[wave][ln][32 + kb + 8 + p];
            }
        }
        #pragma unroll
        for (int nt = 0; nt < 4; ++nt) {
            v16h b0, b1;
            const int dcol = nt * 16 + ln;   // B lane = output-dim column
            const int kb   = 16 * hi;
            #pragma unroll
            for (int p = 0; p < 16; ++p) {
                b0[p] = Vs[cur][kb + p][dcol];         // keys 0..31
                b1[p] = Vs[cur][32 + kb + p][dcol];    // keys 32..63
            }
            oacc[nt] = wmma16x16x32(pa0, b0, oacc[nt]);
            oacc[nt] = wmma16x16x32(pa1, b1, oacc[nt]);
        }
    }

    // ---- epilogue: normalize, store f16 [8192][64] -------------------------
    #pragma unroll
    for (int r = 0; r < 8; ++r) {
        const float inv = 1.0f / lsum[r];
        const size_t row = (size_t)(qbase + r + 8 * hi) * DHEAD;
        #pragma unroll
        for (int nt = 0; nt < 4; ++nt)
            O[row + nt * 16 + ln] = (_Float16)(oacc[nt][r] * inv);
    }
}

// ---------------------------------------------------------------------------
// Kernel 3: output projection, out[b,o,h,w] = sum_c O[n,c]*Wo[o,c] + bo[o]
// ---------------------------------------------------------------------------
__global__ __launch_bounds__(256) void out_proj_kernel(
    const _Float16* __restrict__ Ows, const float* __restrict__ Wo,
    const float* __restrict__ bo, float* __restrict__ out)
{
    __shared__ _Float16 Xs[64][72];
    __shared__ _Float16 Wsh[64][72];

    const int rt = blockIdx.x, ct = blockIdx.y;
    const int n0 = rt * 64, o0 = ct * 64;
    const int b  = n0 >> 10;
    const int s0 = n0 & 1023;

    const int tid  = threadIdx.x;
    const int lane = tid & 31;
    const int wave = tid >> 5;
    const int hi   = lane >> 4;
    const int ln   = lane & 15;
    const int mrow = (wave & 3) * 16;
    const int nt0  = (wave >> 2) * 2;

    v8f acc0 = {}; v8f acc1 = {};

    for (int kc = 0; kc < C_DIM; kc += 64) {
        __syncthreads();
        {   // O chunk (already f16): Xs[i][j] = Ows[(n0+i)*256 + kc + j]
            const int i  = tid >> 2;
            const int jb = (tid & 3) * 16;
            const _Float16* src = Ows + (size_t)(n0 + i) * C_DIM + kc + jb;
            #pragma unroll
            for (int q = 0; q < 16; ++q) Xs[i][jb + q] = src[q];
            const float* wsrc = Wo + (size_t)(o0 + i) * C_DIM + kc + jb;
            #pragma unroll
            for (int q = 0; q < 16; ++q) Wsh[i][jb + q] = (_Float16)wsrc[q];
        }
        __syncthreads();
        #pragma unroll
        for (int ks = 0; ks < 2; ++ks) {
            v16h a;
            const int arow = mrow + ln;
            const int kb   = ks * 32 + 16 * hi;
            #pragma unroll
            for (int p = 0; p < 8; ++p) {
                a[p]     = Xs[arow][kb + p];
                a[8 + p] = Xs[arow][kb + 8 + p];
            }
            v16h b0, b1;
            const int c0  = nt0 * 16 + ln;
            const int kb2 = ks * 32 + 16 * hi;
            #pragma unroll
            for (int p = 0; p < 16; ++p) {
                b0[p] = Wsh[c0][kb2 + p];
                b1[p] = Wsh[c0 + 16][kb2 + p];
            }
            acc0 = wmma16x16x32(a, b0, acc0);
            acc1 = wmma16x16x32(a, b1, acc1);
        }
    }
    const int oc0 = o0 + nt0 * 16 + ln;
    const float bi0 = bo[oc0], bi1 = bo[oc0 + 16];
    #pragma unroll
    for (int r = 0; r < 8; ++r) {
        const int s = s0 + mrow + r + 8 * hi;
        out[((size_t)b * C_DIM + oc0)      * HW + s] = acc0[r] + bi0;
        out[((size_t)b * C_DIM + oc0 + 16) * HW + s] = acc1[r] + bi1;
    }
}

// ---------------------------------------------------------------------------
extern "C" void kernel_launch(void* const* d_in, const int* in_sizes, int n_in,
                              void* d_out, int out_size, void* d_ws, size_t ws_size,
                              hipStream_t stream)
{
    (void)in_sizes; (void)n_in; (void)out_size; (void)ws_size;
    const float* x  = (const float*)d_in[0];
    const float* Wq = (const float*)d_in[1];
    const float* bq = (const float*)d_in[2];
    const float* Wk = (const float*)d_in[3];
    const float* bk = (const float*)d_in[4];
    const float* Wv = (const float*)d_in[5];
    const float* bv = (const float*)d_in[6];
    const float* Wo = (const float*)d_in[7];
    const float* bo = (const float*)d_in[8];
    float* out = (float*)d_out;

    _Float16* Qws = (_Float16*)d_ws;                 // [8192][64]
    _Float16* Kws = Qws + (size_t)NROWS * DHEAD;
    _Float16* Vws = Kws + (size_t)NROWS * DHEAD;
    _Float16* Ows = Vws + (size_t)NROWS * DHEAD;

    proj_qkv_kernel<<<dim3(N_SPATIAL / 64, C_DIM / 64, 3), 256, 0, stream>>>(
        x, Wq, bq, Wk, bk, Wv, bv, Qws, Kws, Vws);

    flash_attn_kernel<<<dim3(NROWS / 64), 128, 0, stream>>>(Qws, Kws, Vws, Ows);

    out_proj_kernel<<<dim3(N_SPATIAL / 64, C_DIM / 64), 256, 0, stream>>>(
        Ows, Wo, bo, out);
}
